// Attention_67989332296336
// MI455X (gfx1250) — matrix operand order repaired and checked by reading, not compiled, not run on hardware
//
#include <hip/hip_runtime.h>
#include <hip/hip_bf16.h>

// ---------------------------------------------------------------------------
// Problem constants (from reference): S=1024, B=32, E=H=A=256, 2H=512
// ---------------------------------------------------------------------------
#define S_DIM 1024
#define B_DIM 32
#define E_DIM 256
#define H_DIM 256
#define A_DIM 256
#define K2H   512

typedef __bf16 v16bf __attribute__((ext_vector_type(16)));
typedef float  v8f   __attribute__((ext_vector_type(8)));
typedef unsigned int v8u __attribute__((ext_vector_type(8)));

__device__ __forceinline__ unsigned short f2bf(float f) {
    unsigned int u = __builtin_bit_cast(unsigned int, f);
    u += 0x7FFFu + ((u >> 16) & 1u);          // round-to-nearest-even
    return (unsigned short)(u >> 16);
}
__device__ __forceinline__ unsigned int pk2(unsigned short lo, unsigned short hi) {
    return (unsigned int)lo | ((unsigned int)hi << 16);
}
__device__ __forceinline__ v8f wmma_bf16(v8u au, v8u bu, v8f acc) {
    v16bf a = __builtin_bit_cast(v16bf, au);
    v16bf b = __builtin_bit_cast(v16bf, bu);
    return __builtin_amdgcn_wmma_f32_16x16x32_bf16(
        /*neg_a=*/false, a, /*neg_b=*/false, b,
        /*c_mod=*/(short)0, acc, /*reuse_a=*/false, /*reuse_b=*/false);
}
// K offset inside a 32-wide K window for A-fragment pair v (ISA 16-bit A 16x32 layout)
__device__ __forceinline__ int ka_off(int v, int half) {
    return ((v < 4) ? 2 * v : 16 + 2 * (v - 4)) + 8 * half;
}
// K offset for B-fragment pair v (same K order as the A fragment)
__device__ __forceinline__ int kb_off(int v, int half) {
    return ((v < 4) ? 2 * v : 16 + 2 * (v - 4)) + 8 * half;
}

// ---------------------------------------------------------------------------
// Prep: x (S,B,E) -> xb[b][t][e] bf16 (row-major, feeds pass-1 B & A tiles)
//       h (S,B,H) -> hbt[b][n][j] bf16 (transposed, feeds pass-3 B tiles
//                                        with two contiguous 16B runs per lane)
//       sq[b][t] = |x_row|^2
// grid (S, B), block E_DIM
// ---------------------------------------------------------------------------
__global__ void prep_kernel(const float* __restrict__ x, const float* __restrict__ h,
                            float* __restrict__ sq,
                            unsigned short* __restrict__ xb,
                            unsigned short* __restrict__ hbt) {
    const int t = blockIdx.x, b = blockIdx.y, e = threadIdx.x;
    const int src = ((t * B_DIM) + b) * E_DIM + e;
    const float xv = x[src];
    xb[((size_t)(b * S_DIM) + t) * E_DIM + e] = f2bf(xv);
    hbt[((size_t)b * H_DIM + e) * S_DIM + t] = f2bf(h[src]);

    __shared__ float red[E_DIM];
    red[e] = xv * xv;
    __syncthreads();
    for (int s = E_DIM / 2; s > 0; s >>= 1) {
        if (e < s) red[e] += red[e + s];
        __syncthreads();
    }
    if (e == 0) sq[b * S_DIM + t] = red[0];
}

// W_g (A, 2H) fp32 -> bf16, same layout. grid 512 x 256
__global__ void wconv_kernel(const float* __restrict__ Wg, unsigned short* __restrict__ wgb) {
    const int i = blockIdx.x * blockDim.x + threadIdx.x;   // exactly A_DIM*K2H threads
    wgb[i] = f2bf(Wg[i]);
}

// ---------------------------------------------------------------------------
// Fused attention: per workgroup = (batch b, 16-query tile t0)
// 128 threads = 4 waves (wave32).
// ---------------------------------------------------------------------------
__global__ void __launch_bounds__(128)
attn_kernel(const unsigned short* __restrict__ xb,   // [B][S][E] bf16
            const unsigned short* __restrict__ hbt,  // [B][H][S] bf16 (transposed)
            const float* __restrict__ sq,            // [B][S]
            const unsigned short* __restrict__ wgb,  // [A][2H] bf16
            const float* __restrict__ h,             // (S,B,H) fp32 (original)
            const float* __restrict__ w_sim, const float* __restrict__ b_sim,
            const float* __restrict__ b_g,
            float* __restrict__ out) {               // (S,B,A) fp32
    __shared__ float          s_scores[16][S_DIM];   // 64 KB: scores -> exp values
    __shared__ unsigned short s_abf[16][S_DIM];      // 32 KB: probabilities (bf16)
    __shared__ unsigned short s_q[16][E_DIM];        //  8 KB: query tile (bf16)
    __shared__ unsigned short s_cat[16][K2H];        // 16 KB: [G | h] row tile (bf16)
    __shared__ float s_red[16][8];
    __shared__ float s_rowmax[16], s_rowsum[16], s_sqq[16];

    const int tid  = threadIdx.x;
    const int wid  = tid >> 5;
    const int lane = tid & 31;
    const int half = lane >> 4;
    const int l15  = lane & 15;

    const int b  = blockIdx.y;
    const int t0 = blockIdx.x * 16;

    const unsigned short* xbb  = xb  + (size_t)b * S_DIM * E_DIM;
    const unsigned short* hbtb = hbt + (size_t)b * H_DIM * S_DIM;
    const float*          sqb  = sq  + (size_t)b * S_DIM;

    const float wsim = w_sim[0];
    const float bsim = b_sim[0];

    // ---- stage query tile + its squared norms -------------------------------
    for (int idx = tid; idx < 16 * E_DIM; idx += 128) {
        const int r = idx >> 8, e = idx & (E_DIM - 1);
        s_q[r][e] = xbb[(size_t)(t0 + r) * E_DIM + e];
    }
    if (tid < 16) s_sqq[tid] = sqb[t0 + tid];
    __syncthreads();

    // ---- pass 1: gram -> distance scores (wave w takes j-tiles w,w+4,...) ---
    for (int jt = wid; jt < S_DIM / 16; jt += 4) {
        const int j0 = jt * 16;
        v8f acc = {};
        for (int e0 = 0; e0 < E_DIM; e0 += 32) {
            v8u au, bu;
#pragma unroll
            for (int v = 0; v < 8; ++v) {
                const int ka = e0 + ka_off(v, half);
                au[v] = *(const unsigned int*)&s_q[l15][ka];
                const int kb = e0 + kb_off(v, half);
                bu[v] = *(const unsigned int*)(xbb + (size_t)(j0 + l15) * E_DIM + kb);
            }
            acc = wmma_bf16(au, bu, acc);
        }
        const float sqj = sqb[j0 + l15];
#pragma unroll
        for (int r = 0; r < 8; ++r) {
            const int m = r + 8 * half;
            const float d2 = s_sqq[m] + sqj - 2.0f * acc[r];
            const float dist = sqrtf(fmaxf(d2, 0.0f) + 1e-12f);
            s_scores[m][j0 + l15] = wsim * dist + bsim;
        }
    }
    __syncthreads();

    // ---- pass 2: softmax over j (row r handled by 8 threads, float4-wide) ---
    {
        const int r = tid >> 3, g = tid & 7;
        float4* row4 = reinterpret_cast<float4*>(&s_scores[r][0]);
        const int vb = g * (S_DIM / 8 / 4), ve = vb + (S_DIM / 8 / 4);  // 32 vec4
        float mx = -3.0e38f;
        for (int i = vb; i < ve; ++i) {
            const float4 t = row4[i];
            mx = fmaxf(mx, fmaxf(fmaxf(t.x, t.y), fmaxf(t.z, t.w)));
        }
        s_red[r][g] = mx;
        __syncthreads();
        if (g == 0) {
            float m2 = s_red[r][0];
#pragma unroll
            for (int i = 1; i < 8; ++i) m2 = fmaxf(m2, s_red[r][i]);
            s_rowmax[r] = m2;
        }
        __syncthreads();
        const float rm = s_rowmax[r];
        float sm = 0.0f;
        for (int i = vb; i < ve; ++i) {
            float4 t = row4[i];
            t.x = __expf(t.x - rm);
            t.y = __expf(t.y - rm);
            t.z = __expf(t.z - rm);
            t.w = __expf(t.w - rm);
            row4[i] = t;                                 // cache exp once
            sm += (t.x + t.y) + (t.z + t.w);
        }
        s_red[r][g] = sm;
        __syncthreads();
        if (g == 0) {
            float s2 = 0.0f;
#pragma unroll
            for (int i = 0; i < 8; ++i) s2 += s_red[r][i];
            s_rowsum[r] = s2;
        }
        __syncthreads();
        const float inv = 1.0f / s_rowsum[r];
        unsigned int* arow = reinterpret_cast<unsigned int*>(&s_abf[r][0]);
        for (int i = vb; i < ve; ++i) {                  // packed bf16 stores
            const float4 t = row4[i];
            arow[2 * i]     = pk2(f2bf(t.x * inv), f2bf(t.y * inv));
            arow[2 * i + 1] = pk2(f2bf(t.z * inv), f2bf(t.w * inv));
        }
    }
    __syncthreads();

    // ---- pass 3: G = A * h  (wave w owns H columns [w*64, w*64+64)) ---------
    for (int nt = 0; nt < 4; ++nt) {
        const int n1 = wid * 64 + nt * 16;
        v8f acc = {};
        for (int k0 = 0; k0 < S_DIM; k0 += 32) {
            v8u au, bu;
#pragma unroll
            for (int v = 0; v < 8; ++v) {
                const int ka = k0 + ka_off(v, half);
                au[v] = *(const unsigned int*)&s_abf[l15][ka];
                const int kb = k0 + kb_off(v, half);
                bu[v] = *(const unsigned int*)(hbtb + (size_t)(n1 + l15) * S_DIM + kb);
            }
            acc = wmma_bf16(au, bu, acc);
        }
#pragma unroll
        for (int r = 0; r < 8; ++r)
            s_cat[r + 8 * half][n1 + l15] = f2bf(acc[r]);
    }
    // fill the h-half of the concat tile (independent of pass-3 results)
    for (int idx = tid; idx < 16 * H_DIM / 2; idx += 128) {
        const int m = idx >> 7, e2 = (idx & 127) * 2;
        const float2 hv =
            *(const float2*)&h[((size_t)(t0 + m) * B_DIM + b) * H_DIM + e2];
        *(unsigned int*)&s_cat[m][H_DIM + e2] = pk2(f2bf(hv.x), f2bf(hv.y));
    }
    __syncthreads();

    // ---- pass 4: Z = tanh(concat(G,h) * W_g^T + b_g) ------------------------
    for (int at = 0; at < 4; ++at) {
        const int a1 = wid * 64 + at * 16;
        v8f acc = {};
        for (int k0 = 0; k0 < K2H; k0 += 32) {
            v8u au, bu;
#pragma unroll
            for (int v = 0; v < 8; ++v) {
                const int ka = k0 + ka_off(v, half);
                au[v] = *(const unsigned int*)&s_cat[l15][ka];
                const int kb = k0 + kb_off(v, half);
                bu[v] = *(const unsigned int*)(wgb + (size_t)(a1 + l15) * K2H + kb);
            }
            acc = wmma_bf16(au, bu, acc);
        }
#pragma unroll
        for (int r = 0; r < 8; ++r) {
            const int m = r + 8 * half;
            const int a = a1 + l15;
            out[((size_t)(t0 + m) * B_DIM + b) * A_DIM + a] = tanhf(acc[r] + b_g[a]);
        }
    }
}

// ---------------------------------------------------------------------------
extern "C" void kernel_launch(void* const* d_in, const int* in_sizes, int n_in,
                              void* d_out, int out_size, void* d_ws, size_t ws_size,
                              hipStream_t stream) {
    const float* x     = (const float*)d_in[0];
    const float* h     = (const float*)d_in[1];
    const float* w_sim = (const float*)d_in[2];
    const float* b_sim = (const float*)d_in[3];
    const float* W_g   = (const float*)d_in[4];
    const float* b_g   = (const float*)d_in[5];
    float* out = (float*)d_out;

    char* ws = (char*)d_ws;
    const size_t XB_BYTES  = (size_t)B_DIM * S_DIM * E_DIM * 2;   // 16 MB
    const size_t HBT_BYTES = (size_t)B_DIM * H_DIM * S_DIM * 2;   // 16 MB
    const size_t SQ_BYTES  = (size_t)B_DIM * S_DIM * 4;           // 128 KB
    unsigned short* xb  = (unsigned short*)(ws);
    unsigned short* hbt = (unsigned short*)(ws + XB_BYTES);
    float*          sq  = (float*)(ws + XB_BYTES + HBT_BYTES);
    unsigned short* wgb = (unsigned short*)(ws + XB_BYTES + HBT_BYTES + SQ_BYTES);

    prep_kernel<<<dim3(S_DIM, B_DIM), E_DIM, 0, stream>>>(x, h, sq, xb, hbt);
    wconv_kernel<<<dim3((A_DIM * K2H) / 256), 256, 0, stream>>>(W_g, wgb);
    attn_kernel<<<dim3(S_DIM / 16, B_DIM), 128, 0, stream>>>(
        xb, hbt, sq, wgb, h, w_sim, b_sim, b_g, out);
}
